// M3GNet_74998718922914
// MI455X (gfx1250) — matrix operand order repaired and checked by reading, not compiled, not run on hardware
//
#include <hip/hip_runtime.h>

typedef __attribute__((ext_vector_type(16))) _Float16 v16h;
typedef __attribute__((ext_vector_type(8)))  float    v8f;

#define NNF 64
#define UNITS 64
#define NEF 64
#define DEG3 9
#define CIN 192
#define CUTOFF 5.0f
#define TB_CUTOFF 4.0f

// Packed-weight layout per branch (halfs): L0 frags (6kc x 4nt) @0, L1 @12288, L2 @16384; branch stride 20480.
#define PW_BRANCH 20480
#define PW_L1OFF  12288
#define PW_L2OFF  16384

__device__ __forceinline__ float siluf(float x) { return x / (1.0f + __expf(-x)); }
__device__ __forceinline__ float sigmf(float x) { return 1.0f / (1.0f + __expf(-x)); }

__device__ __forceinline__ int kmap(int lane, int h) {
  // 16-bit matrix fragment K mapping (contiguous): lanes 0-15 hold K{0..15}, lanes 16-31 hold K{16..31}
  return h + ((lane >> 4) << 4);
}

__device__ __forceinline__ float sincf_(float x) {
  const float PI = 3.14159265358979323846f;
  float ax = fabsf(x);
  if (ax < 1e-8f) return 1.0f;
  return __sinf(PI * x) / (PI * x);
}

__device__ __forceinline__ void smooth_sbf(float r, float out[3]) {
  const float PI = 3.14159265358979323846f;
  const float rc15 = 11.180339887498949f; // 5^1.5
  float f[3];
#pragma unroll
  for (int n = 0; n < 3; ++n) {
    float n1 = (float)(n + 1), n2 = (float)(n + 2);
    float c = ((n & 1) ? -1.0f : 1.0f) * 1.4142135623730951f * PI / rc15 * n1 * n2 *
              rsqrtf(n1 * n1 + n2 * n2);
    f[n] = c * (sincf_(r * n1 / CUTOFF) + sincf_(r * n2 / CUTOFF));
  }
  const float en1 = 9.0f / 65.0f, en2 = 64.0f / 325.0f;
  out[0] = f[0];
  float d1 = 1.0f - en1;
  out[1] = (f[1] + sqrtf(en1) * out[0]) * rsqrtf(d1);
  float d2 = 1.0f - en2 / d1;
  out[2] = (f[2] + sqrtf(en2 / d1) * out[1]) * rsqrtf(d2);
}

__device__ __forceinline__ float poly_cut(float r, float rc) {
  float x = r / rc;
  float x2 = x * x, x3 = x2 * x;
  return (r <= rc) ? (1.0f - 6.0f * x3 * x2 + 15.0f * x2 * x2 - 10.0f * x3) : 0.0f;
}

__device__ __forceinline__ v16h load_frag16(const _Float16* base, int khalfs) {
  union { v16h v; uint4 u[2]; } a;
  const uint4* p = (const uint4*)(base + khalfs);
  a.u[0] = p[0];
  a.u[1] = p[1];
  return a.v;
}

__device__ __forceinline__ v8f wmma16(v16h a, v16h b, v8f c) {
  return __builtin_amdgcn_wmma_f32_16x16x32_f16(false, a, false, b, (short)0, c, false, false);
}

// ---------------- weight packing: f32 [K,64] -> f16 WMMA B-fragments ----------------
struct PackArgs { const float* w[12]; };

__global__ void pack_weights_k(PackArgs pa, _Float16* pw) {
  int id = blockIdx.x * blockDim.x + threadIdx.x;
  if (id >= 4 * PW_BRANCH) return;
  int b = id / PW_BRANCH;
  int r = id - b * PW_BRANCH;
  int L, kc, nt, rem;
  if (r < PW_L1OFF)      { L = 0; int f = r / 512;               kc = f >> 2; nt = f & 3; rem = r & 511; }
  else if (r < PW_L2OFF) { L = 1; int f = (r - PW_L1OFF) / 512;  kc = f >> 2; nt = f & 3; rem = (r - PW_L1OFF) & 511; }
  else                   { L = 2; int f = (r - PW_L2OFF) / 512;  kc = f >> 2; nt = f & 3; rem = (r - PW_L2OFF) & 511; }
  int lane = rem >> 4, h = rem & 15;
  int k = kc * 32 + kmap(lane, h);
  int col = nt * 16 + (lane & 15);
  const float* W = pa.w[b * 3 + L];
  pw[id] = (_Float16)W[k * 64 + col];
}

// ---------------- geometry / embeddings ----------------
__global__ void edge_geom_k(const float* bond_vec, float* bd, float* tcut, float* rbf, int E) {
  int e = blockIdx.x * blockDim.x + threadIdx.x;
  if (e >= E) return;
  float x = bond_vec[e * 3 + 0], y = bond_vec[e * 3 + 1], z = bond_vec[e * 3 + 2];
  float r = sqrtf(x * x + y * y + z * z);
  bd[e] = r;
  tcut[e] = poly_cut(r, TB_CUTOFF);
  float g[3];
  smooth_sbf(r, g);
  rbf[e * 3 + 0] = g[0];
  rbf[e * 3 + 1] = g[1];
  rbf[e * 3 + 2] = g[2];
}

__global__ void edge_embed_k(const float* rbf, const float* Weemb, const float* beemb,
                             float* edge_feat, int E) {
  int id = blockIdx.x * blockDim.x + threadIdx.x;
  if (id >= E * 64) return;
  int e = id >> 6, n = id & 63;
  float s = beemb[n];
#pragma unroll
  for (int k = 0; k < 3; ++k) s += rbf[e * 3 + k] * Weemb[k * 64 + n];
  edge_feat[id] = siluf(s);
}

__global__ void node_embed_k(const float* node_attr, const float* Wnemb, const float* bnemb,
                             float* node_feat, int N) {
  int id = blockIdx.x * blockDim.x + threadIdx.x;
  if (id >= N * 64) return;
  int nd = id >> 6, c = id & 63;
  float s = bnemb[c];
#pragma unroll
  for (int k = 0; k < 16; ++k) s += node_attr[nd * 16 + k] * Wnemb[k * 64 + c];
  node_feat[id] = siluf(s);
}

// ---------------- 3-body basis (block-invariant) ----------------
__global__ void basis_k(const float* bond_vec, const float* bd, const float* tcut,
                        const int* ti, const int* tj, float* basis_w, int T) {
  int t = blockIdx.x * blockDim.x + threadIdx.x;
  if (t >= T) return;
  int i = ti[t], j = tj[t];
  float ax = bond_vec[i * 3 + 0], ay = bond_vec[i * 3 + 1], az = bond_vec[i * 3 + 2];
  float bx = bond_vec[j * 3 + 0], by = bond_vec[j * 3 + 1], bz = bond_vec[j * 3 + 2];
  float ni = sqrtf(ax * ax + ay * ay + az * az);
  float nj = sqrtf(bx * bx + by * by + bz * bz);
  float ct = (ax * bx + ay * by + az * bz) / (ni * nj);
  ct = fminf(1.0f, fmaxf(-1.0f, ct));
  float sb[3];
  smooth_sbf(bd[j], sb);
  float Y[3];
  Y[0] = 0.28209479177387814f;                       // sqrt(1/4pi)
  Y[1] = 0.48860251190291992f * ct;                  // sqrt(3/4pi)*cos
  Y[2] = 0.31539156525252005f * (3.0f * ct * ct - 1.0f);
  float w = tcut[i] * tcut[j];
#pragma unroll
  for (int n = 0; n < 3; ++n)
#pragma unroll
    for (int l = 0; l < 3; ++l)
      basis_w[t * 9 + n * 3 + l] = sb[n] * Y[l] * w;
}

// ---------------- per-block 3-body path ----------------
__global__ void atoms_k(const float* node_feat, const float* Watom3, const float* batom3,
                        float* atoms, int N, int blk) {
  int id = blockIdx.x * blockDim.x + threadIdx.x;
  if (id >= N * 9) return;
  int nd = id / 9, d = id - nd * 9;
  float s = batom3[blk * 9 + d];
  const float* W = Watom3 + blk * 64 * 9;
#pragma unroll 8
  for (int k = 0; k < 64; ++k) s += node_feat[nd * 64 + k] * W[k * 9 + d];
  atoms[id] = sigmf(s);
}

__global__ void scatter3_k(const float* basis_w, const float* atoms, const int* ti, const int* tj,
                           const int* edge_dst, float* nb, int T) {
  int t = blockIdx.x * blockDim.x + threadIdx.x;
  if (t >= T) return;
  int i = ti[t], j = tj[t];
  int d = edge_dst[j];
#pragma unroll
  for (int k = 0; k < 9; ++k)
    atomicAdd(&nb[i * 9 + k], basis_w[t * 9 + k] * atoms[d * 9 + k]);
}

__global__ void edge3_k(const float* nb, const float* W3m, const float* W3g,
                        float* edge_feat, int E, int blk) {
  __shared__ float snb[4][9];
  int el = threadIdx.x >> 6, n = threadIdx.x & 63;
  int e = blockIdx.x * 4 + el;
  if (n < 9) snb[el][n] = (e < E) ? nb[e * 9 + n] : 0.0f;
  __syncthreads();
  if (e >= E) return;
  const float* Wm = W3m + blk * 9 * 64;
  const float* Wg = W3g + blk * 9 * 64;
  float sm = 0.0f, sg = 0.0f;
#pragma unroll
  for (int k = 0; k < 9; ++k) {
    float v = snb[el][k];
    sm += v * Wm[k * 64 + n];
    sg += v * Wg[k * 64 + n];
  }
  edge_feat[e * 64 + n] += siluf(sm) * sigmf(sg);
}

// ---------------- the WMMA gated-MLP kernel (MODE 0: edge residual, MODE 1: node message) --------
template <int MODE>
__global__ __launch_bounds__(128) void gated_mlp_k(
    const float* __restrict__ node_feat, float* __restrict__ edge_feat,
    float* __restrict__ node_accum,
    const int* __restrict__ edge_src, const int* __restrict__ edge_dst,
    const _Float16* __restrict__ pwM, const _Float16* __restrict__ pwG,
    const float* __restrict__ bm, const float* __restrict__ bg,
    const float* __restrict__ rbf, const float* __restrict__ Wmix, int E) {
  __shared__ _Float16 sh_cat[4][16 * CIN];
  __shared__ _Float16 sh_am[4][16 * 64];
  __shared__ _Float16 sh_ag[4][16 * 64];
  __shared__ float    sh_rbf[4][16 * 3];
  __shared__ int      sh_src[4][16];
  __shared__ int      sh_dst[4][16];

  const int w = threadIdx.x >> 5;
  const int lane = threadIdx.x & 31;
  const int lane15 = lane & 15;
  const int hi = (lane >> 4) << 3;          // row/K half select (0 or 8)
  const int tile = blockIdx.x * 4 + w;
  const int eb = tile * 16;

  // stage per-row indices + rbf
  if (lane < 16) {
    int e = eb + lane; if (e >= E) e = E - 1;
    sh_src[w][lane] = edge_src[e] * 64;
    sh_dst[w][lane] = edge_dst[e] * 64;
    sh_rbf[w][lane * 3 + 0] = rbf[e * 3 + 0];
    sh_rbf[w][lane * 3 + 1] = rbf[e * 3 + 1];
    sh_rbf[w][lane * 3 + 2] = rbf[e * 3 + 2];
  }
  __syncthreads();

  // stage cat = [node[src] | node[dst] | edge_feat] as f16 tile [16][192]
  for (int idx = lane; idx < 16 * 48; idx += 32) {
    int row = idx / 48, q = idx - row * 48;
    int e = eb + row; if (e >= E) e = E - 1;
    int c4 = q * 4;
    const float* sp;
    if (q < 16)       sp = node_feat + sh_src[w][row] + c4;
    else if (q < 32)  sp = node_feat + sh_dst[w][row] + (c4 - 64);
    else              sp = edge_feat + e * 64 + (c4 - 128);
    float4 v = *(const float4*)sp;
    int o = row * CIN + c4;
    sh_cat[w][o + 0] = (_Float16)v.x;
    sh_cat[w][o + 1] = (_Float16)v.y;
    sh_cat[w][o + 2] = (_Float16)v.z;
    sh_cat[w][o + 3] = (_Float16)v.w;
  }
  __syncthreads();

  v8f accM[4], accG[4];

  // ---- layer 0: [16,192] x [192,64], shared A for both branches ----
#pragma unroll
  for (int nt = 0; nt < 4; ++nt) {
    float bM = bm[nt * 16 + lane15];
    float bG = bg[nt * 16 + lane15];
#pragma unroll
    for (int r = 0; r < 8; ++r) { accM[nt][r] = bM; accG[nt][r] = bG; }
  }
#pragma unroll
  for (int kc = 0; kc < 6; ++kc) {
    v16h a = load_frag16(&sh_cat[w][lane15 * CIN], kc * 32 + 2 * hi);
#pragma unroll
    for (int nt = 0; nt < 4; ++nt) {
      v16h bMf = load_frag16(pwM, (kc * 4 + nt) * 512 + lane * 16);
      accM[nt] = wmma16(a, bMf, accM[nt]);
      v16h bGf = load_frag16(pwG, (kc * 4 + nt) * 512 + lane * 16);
      accG[nt] = wmma16(a, bGf, accG[nt]);
    }
  }
  __syncthreads();
#pragma unroll
  for (int nt = 0; nt < 4; ++nt)
#pragma unroll
    for (int r = 0; r < 8; ++r) {
      int mrow = r + hi;
      int o = mrow * 64 + nt * 16 + lane15;
      sh_am[w][o] = (_Float16)siluf(accM[nt][r]);
      sh_ag[w][o] = (_Float16)siluf(accG[nt][r]);
    }
  __syncthreads();

  // ---- layers 1,2: [16,64] x [64,64], separate A per branch ----
#pragma unroll
  for (int L = 1; L < 3; ++L) {
    int woff = (L == 1) ? PW_L1OFF : PW_L2OFF;
#pragma unroll
    for (int nt = 0; nt < 4; ++nt) {
      float bM = bm[L * 64 + nt * 16 + lane15];
      float bG = bg[L * 64 + nt * 16 + lane15];
#pragma unroll
      for (int r = 0; r < 8; ++r) { accM[nt][r] = bM; accG[nt][r] = bG; }
    }
#pragma unroll
    for (int kc = 0; kc < 2; ++kc) {
      v16h aM = load_frag16(&sh_am[w][lane15 * 64], kc * 32 + 2 * hi);
      v16h aG = load_frag16(&sh_ag[w][lane15 * 64], kc * 32 + 2 * hi);
#pragma unroll
      for (int nt = 0; nt < 4; ++nt) {
        v16h bMf = load_frag16(pwM, woff + (kc * 4 + nt) * 512 + lane * 16);
        accM[nt] = wmma16(aM, bMf, accM[nt]);
        v16h bGf = load_frag16(pwG, woff + (kc * 4 + nt) * 512 + lane * 16);
        accG[nt] = wmma16(aG, bGf, accG[nt]);
      }
    }
    __syncthreads();
    if (L < 2) {
#pragma unroll
      for (int nt = 0; nt < 4; ++nt)
#pragma unroll
        for (int r = 0; r < 8; ++r) {
          int mrow = r + hi;
          int o = mrow * 64 + nt * 16 + lane15;
          sh_am[w][o] = (_Float16)siluf(accM[nt][r]);
          sh_ag[w][o] = (_Float16)siluf(accG[nt][r]);
        }
      __syncthreads();
    }
  }

  // ---- writeback: out = silu(m) * sigmoid(g) * (rbf @ Wmix) ----
#pragma unroll
  for (int nt = 0; nt < 4; ++nt)
#pragma unroll
    for (int r = 0; r < 8; ++r) {
      int mrow = r + hi;
      int e = eb + mrow;
      int n = nt * 16 + lane15;
      float mix = sh_rbf[w][mrow * 3 + 0] * Wmix[n] +
                  sh_rbf[w][mrow * 3 + 1] * Wmix[64 + n] +
                  sh_rbf[w][mrow * 3 + 2] * Wmix[128 + n];
      float val = siluf(accM[nt][r]) * sigmf(accG[nt][r]) * mix;
      if (e < E) {
        if (MODE == 0) edge_feat[e * 64 + n] += val;
        else atomicAdd(&node_accum[sh_dst[w][mrow] + n], val);
      }
    }
}

__global__ void node_add_k(float* node_feat, const float* nacc, int n) {
  int id = blockIdx.x * blockDim.x + threadIdx.x;
  if (id < n) node_feat[id] += nacc[id];
}

// ---------------- readout ----------------
__global__ void readout1_k(const float* node_feat, const int* gid, float* gsum, float* gcnt, int N) {
  int id = blockIdx.x * blockDim.x + threadIdx.x;
  if (id >= N * 64) return;
  int nd = id >> 6, c = id & 63;
  int g = gid[nd];
  atomicAdd(&gsum[g * 64 + c], node_feat[id]);
  if (c == 0) atomicAdd(&gcnt[g], 1.0f);
}

__global__ void readout2_k(const float* gsum, const float* gcnt,
                           const float* Wf0, const float* bf0,
                           const float* Wf1, const float* bf1,
                           const float* Wf2, const float* bf2, float* out) {
  __shared__ float nv[64], h1[64], h2[64];
  int j = threadIdx.x;
  for (int b = 0; b < 16; ++b) {
    nv[j] = gsum[b * 64 + j] / fmaxf(gcnt[b], 1.0f);
    __syncthreads();
    float s = bf0[j];
    for (int k = 0; k < 64; ++k) s += nv[k] * Wf0[k * 64 + j];
    h1[j] = siluf(s);
    __syncthreads();
    s = bf1[j];
    for (int k = 0; k < 64; ++k) s += h1[k] * Wf1[k * 64 + j];
    h2[j] = siluf(s);
    __syncthreads();
    if (j == 0) {
      float o = bf2[0];
      for (int k = 0; k < 64; ++k) o += h2[k] * Wf2[k];
      out[b] = o;
    }
    __syncthreads();
  }
}

extern "C" void kernel_launch(void* const* d_in, const int* in_sizes, int n_in,
                              void* d_out, int out_size, void* d_ws, size_t ws_size,
                              hipStream_t stream) {
  const float* node_attr = (const float*)d_in[0];
  const float* bond_vec  = (const float*)d_in[1];
  const float* W_nemb = (const float*)d_in[2];
  const float* b_nemb = (const float*)d_in[3];
  const float* W_eemb = (const float*)d_in[4];
  const float* b_eemb = (const float*)d_in[5];
  const float* W_atom3 = (const float*)d_in[6];
  const float* b_atom3 = (const float*)d_in[7];
  const float* W3_main = (const float*)d_in[8];
  const float* W3_gate = (const float*)d_in[9];
  const float* Wem0 = (const float*)d_in[10];
  const float* Wem1 = (const float*)d_in[11];
  const float* Wem2 = (const float*)d_in[12];
  const float* bem  = (const float*)d_in[13];
  const float* Weg0 = (const float*)d_in[14];
  const float* Weg1 = (const float*)d_in[15];
  const float* Weg2 = (const float*)d_in[16];
  const float* beg  = (const float*)d_in[17];
  const float* W_ew = (const float*)d_in[18];
  const float* Wnm0 = (const float*)d_in[19];
  const float* Wnm1 = (const float*)d_in[20];
  const float* Wnm2 = (const float*)d_in[21];
  const float* bnm  = (const float*)d_in[22];
  const float* Wng0 = (const float*)d_in[23];
  const float* Wng1 = (const float*)d_in[24];
  const float* Wng2 = (const float*)d_in[25];
  const float* bng  = (const float*)d_in[26];
  const float* W_nw = (const float*)d_in[27];
  const float* Wf0 = (const float*)d_in[28];
  const float* bf0 = (const float*)d_in[29];
  const float* Wf1 = (const float*)d_in[30];
  const float* bf1 = (const float*)d_in[31];
  const float* Wf2 = (const float*)d_in[32];
  const float* bf2 = (const float*)d_in[33];
  const int* edge_src = (const int*)d_in[34];
  const int* edge_dst = (const int*)d_in[35];
  const int* triple_i = (const int*)d_in[36];
  const int* triple_j = (const int*)d_in[37];
  const int* gid      = (const int*)d_in[38];
  float* out = (float*)d_out;

  const int N = in_sizes[0] / 16;
  const int E = in_sizes[34];
  const int T = in_sizes[36];

  // workspace carve-up
  auto al = [](size_t x) { return (x + 255) & ~(size_t)255; };
  char* w = (char*)d_ws;
  size_t off = 0;
  float* bd       = (float*)(w + off); off = al(off + (size_t)E * 4);
  float* tcut     = (float*)(w + off); off = al(off + (size_t)E * 4);
  float* rbf      = (float*)(w + off); off = al(off + (size_t)E * 3 * 4);
  float* edge_feat= (float*)(w + off); off = al(off + (size_t)E * 64 * 4);
  float* node_feat= (float*)(w + off); off = al(off + (size_t)N * 64 * 4);
  float* nacc     = (float*)(w + off); off = al(off + (size_t)N * 64 * 4);
  float* atoms    = (float*)(w + off); off = al(off + (size_t)N * 9 * 4);
  float* nb       = (float*)(w + off); off = al(off + (size_t)E * 9 * 4);
  float* basis_w  = (float*)(w + off); off = al(off + (size_t)T * 9 * 4);
  float* gsum     = (float*)(w + off); off = al(off + (size_t)16 * 64 * 4);
  float* gcnt     = (float*)(w + off); off = al(off + (size_t)16 * 4);
  _Float16* pw    = (_Float16*)(w + off); off = al(off + (size_t)4 * PW_BRANCH * 2);
  (void)ws_size;

  // pack MLP weights into WMMA f16 B-fragments (block-invariant)
  PackArgs pa;
  pa.w[0] = Wem0; pa.w[1] = Wem1; pa.w[2] = Wem2;
  pa.w[3] = Weg0; pa.w[4] = Weg1; pa.w[5] = Weg2;
  pa.w[6] = Wnm0; pa.w[7] = Wnm1; pa.w[8] = Wnm2;
  pa.w[9] = Wng0; pa.w[10] = Wng1; pa.w[11] = Wng2;
  pack_weights_k<<<(4 * PW_BRANCH + 255) / 256, 256, 0, stream>>>(pa, pw);

  edge_geom_k<<<(E + 255) / 256, 256, 0, stream>>>(bond_vec, bd, tcut, rbf, E);
  edge_embed_k<<<((size_t)E * 64 + 255) / 256, 256, 0, stream>>>(rbf, W_eemb, b_eemb, edge_feat, E);
  node_embed_k<<<((size_t)N * 64 + 255) / 256, 256, 0, stream>>>(node_attr, W_nemb, b_nemb, node_feat, N);
  basis_k<<<(T + 255) / 256, 256, 0, stream>>>(bond_vec, bd, tcut, triple_i, triple_j, basis_w, T);

  const _Float16* pwEM = pw;                  // edge main
  const _Float16* pwEG = pw + PW_BRANCH;      // edge gate
  const _Float16* pwNM = pw + 2 * PW_BRANCH;  // node main
  const _Float16* pwNG = pw + 3 * PW_BRANCH;  // node gate

  int tiles = (E + 15) / 16;
  int mlp_blocks = (tiles + 3) / 4;

  for (int blk = 0; blk < 3; ++blk) {
    atoms_k<<<((size_t)N * 9 + 255) / 256, 256, 0, stream>>>(node_feat, W_atom3, b_atom3, atoms, N, blk);
    hipMemsetAsync(nb, 0, (size_t)E * 9 * 4, stream);
    scatter3_k<<<(T + 255) / 256, 256, 0, stream>>>(basis_w, atoms, triple_i, triple_j, edge_dst, nb, T);
    edge3_k<<<(E + 3) / 4, 256, 0, stream>>>(nb, W3_main, W3_gate, edge_feat, E, blk);
    gated_mlp_k<0><<<mlp_blocks, 128, 0, stream>>>(node_feat, edge_feat, nacc, edge_src, edge_dst,
                                                   pwEM, pwEG, bem, beg, rbf, W_ew, E);
    hipMemsetAsync(nacc, 0, (size_t)N * 64 * 4, stream);
    gated_mlp_k<1><<<mlp_blocks, 128, 0, stream>>>(node_feat, edge_feat, nacc, edge_src, edge_dst,
                                                   pwNM, pwNG, bnm, bng, rbf, W_nw, E);
    node_add_k<<<((size_t)N * 64 + 255) / 256, 256, 0, stream>>>(node_feat, nacc, N * 64);
  }

  hipMemsetAsync(gsum, 0, 16 * 64 * 4, stream);
  hipMemsetAsync(gcnt, 0, 16 * 4, stream);
  readout1_k<<<((size_t)N * 64 + 255) / 256, 256, 0, stream>>>(node_feat, gid, gsum, gcnt, N);
  readout2_k<<<1, 64, 0, stream>>>(gsum, gcnt, Wf0, bf0, Wf1, bf1, Wf2, bf2, out);
  (void)out_size; (void)n_in;
}
